// NegGradOut_61503931678825
// MI455X (gfx1250) — matrix operand order repaired and checked by reading, not compiled, not run on hardware
//
#include <hip/hip_runtime.h>

typedef __attribute__((ext_vector_type(16))) _Float16 v16h;
typedef __attribute__((ext_vector_type(8)))  _Float16 v8h;
typedef __attribute__((ext_vector_type(8)))  float    v8f;
typedef __attribute__((ext_vector_type(4)))  float    f4;

#define W1T_STRIDE 136   // halves; 272B rows -> 16B aligned, banks spread (17n mod 16 bijective)
#define RED_STRIDE 36    // floats per lane row in sRed; 144B -> 16B aligned, banks spread

__global__ __launch_bounds__(128) void zero_f32_kernel(float* __restrict__ p, int n) {
    int i = blockIdx.x * 128 + threadIdx.x;
    if (i < n) p[i] = 0.0f;
}

__global__ __launch_bounds__(128) __attribute__((amdgpu_num_vgpr(256)))
void energy_force_kernel(
    const float* __restrict__ coord,       // [N,3]
    const float* __restrict__ feat_base,   // [N,128]
    const float* __restrict__ W_feat,      // [3,128]
    const float* __restrict__ W1,          // [128,64]
    const float* __restrict__ b1,          // [64]
    const float* __restrict__ W2,          // [64,1]
    const float* __restrict__ b2,          // [1]
    const int*   __restrict__ batch_index, // [N]
    float* __restrict__ res,               // [B] (pre-zeroed)
    float* __restrict__ ngrad,             // [N,3]
    int N, int nTiles)
{
    // ---- LDS (~38 KB) ----
    __shared__ _Float16 sW1T[64 * W1T_STRIDE];   // W1^T[h][d] f16 (forward B operand)
    __shared__ float    sM[3 * 64];              // M[c][h] = sum_d W_feat[c,d]*W1[d,h] (f32)
    __shared__ float    sB1[64];
    __shared__ float    sW2[64];
    __shared__ float    sRed[4][32 * RED_STRIDE];// per-wave reduction scratch
    __shared__ float    sPe[4][16];              // per-wave atom energies
    __shared__ float    sGr[4][48];              // per-wave grad tile [16][3]

    const int tid = threadIdx.x;
    for (int i = tid; i < 128 * 64; i += 128) {
        int d = i >> 6, j = i & 63;
        sW1T[j * W1T_STRIDE + d] = (_Float16)W1[i];
    }
    if (tid < 64) { sB1[tid] = b1[tid]; sW2[tid] = W2[tid]; }
    // M = W_feat @ W1, full f32: fuses the forward coord term AND the whole
    // backward chain (g @ W1^T) @ W_feat^T (same 3x64 matrix).
    for (int e = tid; e < 192; e += 128) {
        const int c = e >> 6;      // 0..2
        const int h = e & 63;
        float s = 0.f;
#pragma unroll 8
        for (int d = 0; d < 128; ++d)
            s += W_feat[c * 128 + d] * W1[d * 64 + h];
        sM[c * 64 + h] = s;
    }
    __syncthreads();

    const int  wave = tid >> 5;
    const int  lane = tid & 31;
    const int  n16  = lane & 15;
    const bool hiH  = lane >= 16;
    const int  kbA  = hiH ? 8  : 0;   // 16-bit A-fragment K base (split runs)
    const int  kbB  = hiH ? 8  : 0;   // 16-bit B-fragment K base (split runs)
    const int  mOff = hiH ? 8 : 0;
    const float b2v = b2[0];

    // ---- per-lane invariants (depend only on n16): h = hc*16 + n16 ----
    float mv0[4], mv1[4], mv2[4], b1v[4], w2v[4];
#pragma unroll
    for (int hc = 0; hc < 4; ++hc) {
        mv0[hc] = sM[       hc * 16 + n16];
        mv1[hc] = sM[ 64 +  hc * 16 + n16];
        mv2[hc] = sM[128 +  hc * 16 + n16];
        b1v[hc] = sB1[hc * 16 + n16];
        w2v[hc] = sW2[hc * 16 + n16];
    }
    // reduction-phase assignment: lane = mA*4 + q  (q: 0=energy, 1..3=grad c)
    const int q  = lane & 3;
    const int mA = lane >> 2;   // 0..7 ; also handles mB = mA + 8

    const int waveStride = gridDim.x * 4;
    for (int tile = blockIdx.x * 4 + wave; tile < nTiles; tile += waveStride) {
        const int atomM = tile * 16 + n16;
        const int atomC = atomM < N ? atomM : N - 1;

        // prefetch next grid-stride tile's feat_base rows (64 x 128B lines)
        {
            const int tnext = tile + waveStride;
            if (tnext < nTiles) {
                int an = tnext * 16 + n16;
                if (an >= N) an = N - 1;
                const float* fbn = feat_base + (long)an * 128 + (hiH ? 32 : 0);
                __builtin_prefetch(fbn, 0, 3);
                __builtin_prefetch(fbn + 64, 0, 3);
            }
        }

        const float cx = coord[atomC * 3 + 0];
        const float cy = coord[atomC * 3 + 1];
        const float cz = coord[atomC * 3 + 2];
        const float* fb = feat_base + (long)atomC * 128;

        // ---- A fragments: pure f32->f16 convert of feat_base ----
        v16h A[4];
#pragma unroll
        for (int cc = 0; cc < 4; ++cc) {
            v16h a;
#pragma unroll
            for (int h2 = 0; h2 < 2; ++h2) {
                const int db = cc * 32 + kbA + h2 * 16;
                f4 f0 = *(const f4*)(fb + db);
                f4 f1 = *(const f4*)(fb + db + 4);
#pragma unroll
                for (int qq = 0; qq < 4; ++qq) {
                    a[h2 * 8 + qq]     = (_Float16)f0[qq];
                    a[h2 * 8 + 4 + qq] = (_Float16)f1[qq];
                }
            }
            A[cc] = a;
        }

        // ---- forward feat_base @ W1: 16 WMMAs, 4 independent accumulators ----
        v8f accF[4];
#pragma unroll
        for (int dc = 0; dc < 4; ++dc) {
            v16h Bf[4];
#pragma unroll
            for (int hc = 0; hc < 4; ++hc) {
                const _Float16* bp = &sW1T[(hc * 16 + n16) * W1T_STRIDE + dc * 32 + kbB];
                v8h blo = *(const v8h*)bp;
                v8h bhi = *(const v8h*)(bp + 16);
                Bf[hc] = __builtin_shufflevector(blo, bhi, 0,1,2,3,4,5,6,7,8,9,10,11,12,13,14,15);
            }
#pragma unroll
            for (int hc = 0; hc < 4; ++hc) {
                if (dc == 0) {
                    v8f zz = {0.f,0.f,0.f,0.f,0.f,0.f,0.f,0.f};
                    accF[hc] = __builtin_amdgcn_wmma_f32_16x16x32_f16(false, A[0], false, Bf[hc],
                                                                      (short)0, zz, false, false);
                } else {
                    accF[hc] = __builtin_amdgcn_wmma_f32_16x16x32_f16(false, A[dc], false, Bf[hc],
                                                                      (short)0, accF[hc], false, false);
                }
            }
        }

        // ---- fused epilogue, r-outer: z = accF + b1 + coord@M ; silu ; pack ----
        // C layout: element r -> (m = mOff + r, h = hc*16 + n16)
        float* const redw = &sRed[wave][0];
#pragma unroll
        for (int r = 0; r < 8; ++r) {
            const float cxr = __shfl(cx, mOff + r);
            const float cyr = __shfl(cy, mOff + r);
            const float czr = __shfl(cz, mOff + r);
            float pe = 0.f, p0 = 0.f, p1 = 0.f, p2 = 0.f;
#pragma unroll
            for (int hc = 0; hc < 4; ++hc) {
                float zz  = accF[hc][r] + b1v[hc]
                          + cxr * mv0[hc] + cyr * mv1[hc] + czr * mv2[hc];
                float sig = 1.0f / (1.0f + __expf(-zz));
                float si  = zz * sig;
                pe += w2v[hc] * si;
                float dsl = sig + si * (1.0f - sig);     // silu'(z)
                float g   = -(w2v[hc] * dsl);            // negated -> -grad
                p0 += g * mv0[hc];                       // -grad[m,c] = sum_h g*M[c,h]
                p1 += g * mv1[hc];
                p2 += g * mv2[hc];
            }
            f4 pk = {pe, p0, p1, p2};
            *(f4*)(redw + lane * RED_STRIDE + r * 4) = pk;   // ds_store_b128
        }

        // ---- cross-lane reduction via LDS transpose (conflict-free loads) ----
        // lane (mA*4+q) sums column (m,q) for m = mA and m = mA+8 over 16 h-lanes
        {
            float t0 = 0.f, t1 = 0.f;
#pragma unroll
            for (int n = 0; n < 16; ++n) {
                t0 += redw[ n        * RED_STRIDE + mA * 4 + q];  // addr = n*36 + lane
                t1 += redw[(n + 16)  * RED_STRIDE + mA * 4 + q];
            }
            if (q == 0) {
                sPe[wave][mA]     = t0;
                sPe[wave][mA + 8] = t1;
            } else {
                sGr[wave][ mA      * 3 + (q - 1)] = t0;
                sGr[wave][(mA + 8) * 3 + (q - 1)] = t1;
            }
        }
        // same-wave LDS ordering: DScnt is in-order; compiler inserts waits

        // ---- energy: merged atomics (batch_index is sorted) ----
        if (n16 == 0) {
            const int a0 = tile * 16 + mOff;
            if (a0 < N) {
                float pev[8];
#pragma unroll
                for (int r = 0; r < 8; ++r) pev[r] = sPe[wave][mOff + r];
                if (a0 + 7 < N) {
                    int bi0 = batch_index[a0];
                    int bi7 = batch_index[a0 + 7];
                    if (bi0 == bi7) {
                        float s = pev[0] + pev[1] + pev[2] + pev[3]
                                + pev[4] + pev[5] + pev[6] + pev[7];
                        atomicAdd(&res[bi0], s + 8.0f * b2v);
                    } else {
#pragma unroll
                        for (int r = 0; r < 8; ++r)
                            atomicAdd(&res[batch_index[a0 + r]], pev[r] + b2v);
                    }
                } else {
#pragma unroll
                    for (int r = 0; r < 8; ++r)
                        if (a0 + r < N)
                            atomicAdd(&res[batch_index[a0 + r]], pev[r] + b2v);
                }
            }
        }

        // ---- coalesced 48-float grad store ----
        const int tbase = tile * 16;
        if (tbase + 16 <= N) {
            if (lane < 12) {   // base = tile*192B -> 16B aligned
                f4 v = *(const f4*)&sGr[wave][lane * 4];
                *(f4*)(ngrad + (long)tbase * 3 + lane * 4) = v;
            }
        } else if (!hiH && atomM < N) {
            long o = (long)atomM * 3;
            ngrad[o + 0] = sGr[wave][n16 * 3 + 0];
            ngrad[o + 1] = sGr[wave][n16 * 3 + 1];
            ngrad[o + 2] = sGr[wave][n16 * 3 + 2];
        }
    }
}

extern "C" void kernel_launch(void* const* d_in, const int* in_sizes, int n_in,
                              void* d_out, int out_size, void* d_ws, size_t ws_size,
                              hipStream_t stream) {
    const float* coord       = (const float*)d_in[0];
    const float* feat_base   = (const float*)d_in[1];
    const float* W_feat      = (const float*)d_in[2];
    const float* W1          = (const float*)d_in[3];
    const float* b1          = (const float*)d_in[4];
    const float* W2          = (const float*)d_in[5];
    const float* b2          = (const float*)d_in[6];
    const int*   batch_index = (const int*)d_in[7];

    const int N  = in_sizes[0] / 3;          // coord is [N,3]
    const int Bg = out_size - 3 * N;         // out = res[B] ++ (-grad)[N,3]

    float* out  = (float*)d_out;
    float* res  = out;
    float* ngrd = out + Bg;

    // res region is accumulated via atomics -> zero it first (stream-ordered)
    zero_f32_kernel<<<(Bg + 127) / 128, 128, 0, stream>>>(res, Bg);

    const int nTiles = (N + 15) / 16;
    int blocks = (nTiles + 3) / 4;
    if (blocks > 1536) blocks = 1536;        // grid-stride persistent waves

    energy_force_kernel<<<blocks, 128, 0, stream>>>(
        coord, feat_base, W_feat, W1, b1, W2, b2, batch_index,
        res, ngrd, N, nTiles);
}
